// FeedForwardNN_13486197310319
// MI455X (gfx1250) — matrix operand order repaired and checked by reading, not compiled, not run on hardware
//
#include <hip/hip_runtime.h>
#include <hip/hip_bf16.h>

typedef __attribute__((ext_vector_type(16))) _Float16 v16h;
typedef __attribute__((ext_vector_type(8)))  _Float16 v8h;
typedef __attribute__((ext_vector_type(8)))  float    v8f;
typedef __attribute__((ext_vector_type(4)))  float    v4f;

#define SEQ_T   2048
#define BATCH_N 1024
#define HID     32
#define GATES   128   // 4*HID

// ---- fast activations: single trans op each (v_tanh_f32 / v_exp_f32+v_rcp_f32)
__device__ __forceinline__ float tanh_fast(float x) {
#if __has_builtin(__builtin_amdgcn_tanhf)
    return __builtin_amdgcn_tanhf(x);                  // v_tanh_f32
#else
    return 2.0f * __builtin_amdgcn_rcpf(1.0f + __expf(-2.0f * x)) - 1.0f;
#endif
}
__device__ __forceinline__ float sigm(float x) {
#if __has_builtin(__builtin_amdgcn_tanhf)
    return 0.5f * __builtin_amdgcn_tanhf(0.5f * x) + 0.5f;
#else
    return __builtin_amdgcn_rcpf(1.0f + __expf(-x));   // v_exp + v_rcp, no div
#endif
}

// One wave (32 lanes) owns a 16-row batch tile for the entire T-scan of one
// LSTM layer. Weights live in WMMA B-fragments (VGPRs) for the whole scan;
// cell state c lives in VGPRs; h round-trips through a 1KB LDS tile to do the
// C-layout -> A-layout transpose needed to feed it back into the next WMMA.
__global__ __launch_bounds__(32) void lstm_layer_kernel(
    const float*    __restrict__ x0,    // layer0 input (T,B,1) fp32, else null
    const _Float16* __restrict__ xin,   // layer>=1 input (T,B,32) f16, else null
    const float*    __restrict__ Wih,   // (128, in_dim) fp32
    const float*    __restrict__ Whh,   // (128, 32) fp32
    const float*    __restrict__ bih,   // (128,)
    const float*    __restrict__ bhh,   // (128,)
    _Float16*       __restrict__ hout,  // (T,B,32) f16
    int layer0)
{
    __shared__ _Float16 WB[GATES * 64];   // [gate n][k] : [Wih | Whh], 16 KB
    __shared__ _Float16 hbuf[16 * HID];   // current h tile, row-major, 1 KB

    const int lane  = threadIdx.x;
    const int lhalf = lane & 15;          // N-column / M-row within half
    const int hi    = lane >> 4;          // which 16-lane half
    const int b0    = blockIdx.x * 16;    // batch tile base

    // ---- Build f16 weight table [Wih | Whh] (N-major => contiguous B frags)
    for (int idx = lane; idx < GATES * 64; idx += 32) {
        int n = idx >> 6, k = idx & 63;
        float w;
        if (k < 32) w = layer0 ? 0.0f : Wih[n * 32 + k];
        else        w = Whh[n * 32 + (k - 32)];
        WB[idx] = (_Float16)w;
    }
    for (int idx = lane; idx < 16 * HID; idx += 32) hbuf[idx] = (_Float16)0.0f;
    __syncthreads();

    // ---- Per-lane bias (and layer0 input-weight column) for accumulator init
    float biasv[8], wxv[8];
#pragma unroll
    for (int nt = 0; nt < 8; ++nt) {
        int c = nt * 16 + lhalf;
        biasv[nt] = bih[c] + bhh[c];
        wxv[nt]   = layer0 ? Wih[c] : 0.0f;   // Wih0 is (128,1)
    }

    // ---- Load WMMA B fragments (stay in VGPRs for all 2048 steps)
    // B 32x16 f16 layout: lane = N%16; lane<16 K 0-7 & 16-23, lane>=16 K 8-15 & 24-31
    v16h bf[2][8];
#pragma unroll
    for (int kt = 0; kt < 2; ++kt)
#pragma unroll
        for (int nt = 0; nt < 8; ++nt) {
            int row = nt * 16 + lhalf;
            v8h blo = *(const v8h*)&WB[row * 64 + kt * 32 + hi * 8];
            v8h bhi = *(const v8h*)&WB[row * 64 + kt * 32 + 16 + hi * 8];
            bf[kt][nt] = __builtin_shufflevector(blo, bhi,
                           0,1,2,3,4,5,6,7,8,9,10,11,12,13,14,15);
        }

    v8f cst0, cst1;                        // cell state, C-tile layout
#pragma unroll
    for (int r = 0; r < 8; ++r) { cst0[r] = 0.0f; cst1[r] = 0.0f; }

    for (int t = 0; t < SEQ_T; ++t) {
        // ---- h A-fragment (16x32 f16): lane<16 row=l, K 0-7 & 16-23;
        //      lane>=16 row=l-16, K 8-15 & 24-31
        v8h hlo = *(const v8h*)&hbuf[lhalf * HID + hi * 8];
        v8h hhi = *(const v8h*)&hbuf[lhalf * HID + 16 + hi * 8];
        v16h a_h = __builtin_shufflevector(hlo, hhi,
                     0,1,2,3,4,5,6,7,8,9,10,11,12,13,14,15);

        v8f acc[8];
        if (layer0) {
            // gates_pre = bias + x[t,b] * Wih0[:,0]  (scalar input dim)
            const float* xp = x0 + (size_t)t * BATCH_N + b0 + hi * 8;
            float xr[8];
            *(v4f*)&xr[0] = *(const v4f*)&xp[0];
            *(v4f*)&xr[4] = *(const v4f*)&xp[4];
#pragma unroll
            for (int nt = 0; nt < 8; ++nt)
#pragma unroll
                for (int r = 0; r < 8; ++r)
                    acc[nt][r] = biasv[nt] + wxv[nt] * xr[r];
        } else {
            const _Float16* xp =
                xin + ((size_t)t * BATCH_N + b0 + lhalf) * HID + hi * 8;
            if (t + 1 < SEQ_T)
                __builtin_prefetch(xp + (size_t)BATCH_N * HID, 0, 0);
            v8h xlo = *(const v8h*)&xp[0];
            v8h xhi = *(const v8h*)&xp[16];
            v16h a_x = __builtin_shufflevector(xlo, xhi,
                         0,1,2,3,4,5,6,7,8,9,10,11,12,13,14,15);
#pragma unroll
            for (int nt = 0; nt < 8; ++nt) {
                v8f cinit;
#pragma unroll
                for (int r = 0; r < 8; ++r) cinit[r] = biasv[nt];
                acc[nt] = __builtin_amdgcn_wmma_f32_16x16x32_f16(
                    false, a_x, false, bf[0][nt], (short)0, cinit, false, false);
            }
        }
#pragma unroll
        for (int nt = 0; nt < 8; ++nt)
            acc[nt] = __builtin_amdgcn_wmma_f32_16x16x32_f16(
                false, a_h, false, bf[1][nt], (short)0, acc[nt], false, false);

        // ---- Elementwise gates (C layout: VGPR r = batch row r+hi*8,
        //      lane = hidden col; nt 0,1=i  2,3=f  4,5=g  6,7=o)
        v8f hh0, hh1;
#pragma unroll
        for (int r = 0; r < 8; ++r) {
            float i0 = sigm(acc[0][r]), f0 = sigm(acc[2][r]);
            float g0 = tanh_fast(acc[4][r]), o0 = sigm(acc[6][r]);
            float c0 = f0 * cst0[r] + i0 * g0;
            cst0[r] = c0;
            hh0[r] = o0 * tanh_fast(c0);

            float i1 = sigm(acc[1][r]), f1 = sigm(acc[3][r]);
            float g1 = tanh_fast(acc[5][r]), o1 = sigm(acc[7][r]);
            float c1 = f1 * cst1[r] + i1 * g1;
            cst1[r] = c1;
            hh1[r] = o1 * tanh_fast(c1);
        }

        // ---- Scatter h into LDS tile (row-major [row][hid]) for next step's
        //      A fragment; DS ops are in-order per wave.
#pragma unroll
        for (int r = 0; r < 8; ++r) {
            int row = hi * 8 + r;
            hbuf[row * HID + lhalf]      = (_Float16)hh0[r];
            hbuf[row * HID + 16 + lhalf] = (_Float16)hh1[r];
        }
        __syncthreads();   // single-wave WG: lowers to wait + s_nop

        // ---- Packed coalesced write of the h tile to global (layer output)
        {
            _Float16* op = hout + ((size_t)t * BATCH_N + b0) * HID;
            v8h p0 = *(const v8h*)&hbuf[lane * 16];
            v8h p1 = *(const v8h*)&hbuf[lane * 16 + 8];
            *(v8h*)&op[lane * 16]     = p0;
            *(v8h*)&op[lane * 16 + 8] = p1;
        }
        __syncthreads();
    }
}

// Final projection: out[t,b] = dot(h[t,b,:], Wfc) + bfc  (memory-bound)
__global__ __launch_bounds__(256) void fc_kernel(
    const _Float16* __restrict__ hin, const float* __restrict__ Wfc,
    const float* __restrict__ bfc, float* __restrict__ out, int n)
{
    int i = blockIdx.x * blockDim.x + threadIdx.x;
    if (i >= n) return;
    const _Float16* hp = hin + (size_t)i * HID;
    float s = bfc[0];
#pragma unroll
    for (int c = 0; c < 4; ++c) {
        v8h v = *(const v8h*)&hp[c * 8];
#pragma unroll
        for (int j = 0; j < 8; ++j) s += (float)v[j] * Wfc[c * 8 + j];
    }
    out[i] = s;
}

extern "C" void kernel_launch(void* const* d_in, const int* in_sizes, int n_in,
                              void* d_out, int out_size, void* d_ws, size_t ws_size,
                              hipStream_t stream) {
    const float* x    = (const float*)d_in[0];
    const float* Wih0 = (const float*)d_in[1];
    const float* Whh0 = (const float*)d_in[2];
    const float* bih0 = (const float*)d_in[3];
    const float* bhh0 = (const float*)d_in[4];
    const float* Wih1 = (const float*)d_in[5];
    const float* Whh1 = (const float*)d_in[6];
    const float* bih1 = (const float*)d_in[7];
    const float* bhh1 = (const float*)d_in[8];
    const float* Wih2 = (const float*)d_in[9];
    const float* Whh2 = (const float*)d_in[10];
    const float* bih2 = (const float*)d_in[11];
    const float* bhh2 = (const float*)d_in[12];
    const float* Wfc  = (const float*)d_in[13];
    const float* bfc  = (const float*)d_in[14];

    // Ping-pong f16 activation buffers in workspace: 2 x 128 MB
    _Float16* bufA = (_Float16*)d_ws;
    _Float16* bufB = bufA + (size_t)SEQ_T * BATCH_N * HID;

    dim3 grid(BATCH_N / 16), block(32);
    lstm_layer_kernel<<<grid, block, 0, stream>>>(x, nullptr, Wih0, Whh0,
                                                  bih0, bhh0, bufA, 1);
    lstm_layer_kernel<<<grid, block, 0, stream>>>(nullptr, bufA, Wih1, Whh1,
                                                  bih1, bhh1, bufB, 0);
    lstm_layer_kernel<<<grid, block, 0, stream>>>(nullptr, bufB, Wih2, Whh2,
                                                  bih2, bhh2, bufA, 0);

    int n = SEQ_T * BATCH_N;
    fc_kernel<<<(n + 255) / 256, 256, 0, stream>>>(bufA, Wfc, bfc,
                                                   (float*)d_out, n);
}
